// CosineVectorQuantizer_876173328854
// MI455X (gfx1250) — matrix operand order repaired and checked by reading, not compiled, not run on hardware
//
#include <hip/hip_runtime.h>
#include <hip/hip_bf16.h>

typedef __attribute__((ext_vector_type(16))) _Float16 v16h;
typedef __attribute__((ext_vector_type(8)))  _Float16 v8h;
typedef __attribute__((ext_vector_type(8)))  float    v8f;

#define N_LAT   32768
#define EDIM    64
#define N_E     8192
#define NTILE   (N_E / 16)            // 512 codebook tiles
#define XQ_OFF  0
#define LOSS_OFF (N_LAT * EDIM)       // 2097152
#define IDX_OFF  (LOSS_OFF + 1)
#define CB_OFF   (IDX_OFF + N_LAT)
#define LOSS_SCALE (1.25f / (float)(N_LAT * EDIM))
#define NORM_EPS 1e-12f

// ---------------------------------------------------------------------------
// Kernel 1: normalize codebook rows -> f16, store in WMMA B-fragment layout.
// Layout per (tile t, kchunk h): 32 lanes x 16 halfs contiguous per lane.
//   lane N      (N=0..15): column N of B, K = h*32 + 0..15  (sequential)
//   lane N+16            : column N of B, K = h*32 + 16..31 (sequential)
// One wave handles one 16-row tile; lane handles half a row (32 f32).
// ---------------------------------------------------------------------------
__global__ __launch_bounds__(128) void vq_prep_codebook(
    const float* __restrict__ emb, _Float16* __restrict__ wsB) {
  int tid  = blockIdx.x * blockDim.x + threadIdx.x;
  int tile = tid >> 5;          // 0..511
  int lane = tid & 31;
  int rloc = lane & 15;
  int h    = lane >> 4;
  int row  = tile * 16 + rloc;

  const float* src = emb + (size_t)row * EDIM + h * 32;
  float vals[32];
  float ss = 0.f;
#pragma unroll
  for (int i = 0; i < 8; ++i) {
    float4 f = reinterpret_cast<const float4*>(src)[i];
    vals[4*i+0] = f.x; vals[4*i+1] = f.y; vals[4*i+2] = f.z; vals[4*i+3] = f.w;
    ss += f.x*f.x + f.y*f.y + f.z*f.z + f.w*f.w;
  }
  float tot = ss + __shfl_xor(ss, 16, 32);     // partner lane holds other half
  float inv = 1.0f / fmaxf(sqrtf(tot), NORM_EPS);

  _Float16* dstBase = wsB + ((size_t)(tile * 2 + h)) * 512;  // 32 lanes * 16 halfs
#pragma unroll
  for (int half = 0; half < 2; ++half) {
    // halfs [0..15] -> dest lane rloc ; halfs [16..31] -> dest lane rloc+16
    _Float16* d = dstBase + (size_t)(rloc + 16 * half) * 16;
#pragma unroll
    for (int i = 0; i < 2; ++i) {
      v8h hv;
#pragma unroll
      for (int j = 0; j < 8; ++j)
        hv[j] = (_Float16)(vals[16*half + 8*i + j] * inv);
      *reinterpret_cast<v8h*>(d + 8 * i) = hv;
    }
  }
}

// ---------------------------------------------------------------------------
// Kernel 2: fused normalize(x) -> f16 WMMA GEMM -> row argmax -> gather.
// 4 waves / block; each wave owns TWO 16-row M tiles (32 rows) and reuses
// every B fragment for two independent WMMA accumulator chains (halves L2
// traffic and fills the WMMA->VALU hazard slots with real work).
// ---------------------------------------------------------------------------
__global__ __launch_bounds__(128) void vq_main(
    const float* __restrict__ x, const float* __restrict__ emb,
    const _Float16* __restrict__ wsB, float* __restrict__ out,
    float* __restrict__ partial) {
  __shared__ _Float16 smem[4 * 32 * 64];      // 4 waves x 32 rows x 64 halfs

  int wave  = threadIdx.x >> 5;
  int lane  = threadIdx.x & 31;
  int rloc  = lane & 15;
  int h     = lane >> 4;
  int wBase = blockIdx.x * 128 + wave * 32;   // first of 32 rows for this wave

  // --- normalize 32 latent rows into LDS (one full row per lane) ---
  {
    int n = wBase + lane;
    const float* src = x + (size_t)n * EDIM;
    float vals[64];
    float ss = 0.f;
#pragma unroll
    for (int i = 0; i < 16; ++i) {
      float4 f = reinterpret_cast<const float4*>(src)[i];
      vals[4*i+0] = f.x; vals[4*i+1] = f.y; vals[4*i+2] = f.z; vals[4*i+3] = f.w;
      ss += f.x*f.x + f.y*f.y + f.z*f.z + f.w*f.w;
    }
    float inv = 1.0f / fmaxf(sqrtf(ss), NORM_EPS);
    _Float16* d = smem + wave * 2048 + lane * 64;
#pragma unroll
    for (int i = 0; i < 8; ++i) {
      v8h hv;
#pragma unroll
      for (int j = 0; j < 8; ++j) hv[j] = (_Float16)(vals[8*i + j] * inv);
      *reinterpret_cast<v8h*>(d + 8 * i) = hv;
    }
  }
  __syncthreads();

  // --- build A fragments for both sub-tiles (16x32 f16 layout):
  //     lane<16 holds K {0..15}, lane>=16 holds K {16..31} ---
  v16h a[2][2];
#pragma unroll
  for (int u = 0; u < 2; ++u) {
    const _Float16* abase =
        smem + wave * 2048 + (u * 16 + rloc) * 64 + ((lane < 16) ? 0 : 16);
    v8h lo0 = *reinterpret_cast<const v8h*>(abase +  0);
    v8h hi0 = *reinterpret_cast<const v8h*>(abase +  8);
    v8h lo1 = *reinterpret_cast<const v8h*>(abase + 32);
    v8h hi1 = *reinterpret_cast<const v8h*>(abase + 40);
    a[u][0] = __builtin_shufflevector(lo0, hi0,
              0,1,2,3,4,5,6,7,8,9,10,11,12,13,14,15);
    a[u][1] = __builtin_shufflevector(lo1, hi1,
              0,1,2,3,4,5,6,7,8,9,10,11,12,13,14,15);
  }

  // --- main loop: 512 tiles; each B fragment feeds TWO WMMA chains ---
  float best[2][8];
  int   bidx[2][8];
#pragma unroll
  for (int u = 0; u < 2; ++u)
#pragma unroll
    for (int g = 0; g < 8; ++g) { best[u][g] = -3.402823e38f; bidx[u][g] = 0; }

  const _Float16* bptr = wsB + (size_t)lane * 16;
  for (int t = 0; t < NTILE; ++t) {
    v16h b0 = *reinterpret_cast<const v16h*>(bptr);
    v16h b1 = *reinterpret_cast<const v16h*>(bptr + 512);
    bptr += 1024;
    v8f c0 = {}, c1 = {};
    c0 = __builtin_amdgcn_wmma_f32_16x16x32_f16(false, a[0][0], false, b0,
                                                (short)0, c0, false, false);
    c1 = __builtin_amdgcn_wmma_f32_16x16x32_f16(false, a[1][0], false, b0,
                                                (short)0, c1, false, false);
    c0 = __builtin_amdgcn_wmma_f32_16x16x32_f16(false, a[0][1], false, b1,
                                                (short)0, c0, false, false);
    c1 = __builtin_amdgcn_wmma_f32_16x16x32_f16(false, a[1][1], false, b1,
                                                (short)0, c1, false, false);
    int cidx = t * 16 + rloc;   // this lane's column = codebook index
#pragma unroll
    for (int g = 0; g < 8; ++g) {
      bool b0g = c0[g] > best[0][g];          // strict > keeps earliest index
      best[0][g] = b0g ? c0[g] : best[0][g];
      bidx[0][g] = b0g ? cidx  : bidx[0][g];
      bool b1g = c1[g] > best[1][g];
      best[1][g] = b1g ? c1[g] : best[1][g];
      bidx[1][g] = b1g ? cidx  : bidx[1][g];
    }
  }

  // --- argmax reduction across the 16 columns (within each lane half) ---
#pragma unroll
  for (int mask = 1; mask < 16; mask <<= 1) {
#pragma unroll
    for (int u = 0; u < 2; ++u) {
#pragma unroll
      for (int g = 0; g < 8; ++g) {
        float ov = __shfl_xor(best[u][g], mask, 16);
        int   oi = __shfl_xor(bidx[u][g], mask, 16);
        bool take = (ov > best[u][g]) || (ov == best[u][g] && oi < bidx[u][g]);
        best[u][g] = take ? ov : best[u][g];
        bidx[u][g] = take ? oi : bidx[u][g];
      }
    }
  }

  // --- gather emb[idx], write x_q_st / codebook_vec / indices, loss partial ---
  float lossAcc = 0.f;
  int k0 = rloc * 4;                 // 16 lanes x 4 floats cover one row
#pragma unroll
  for (int u = 0; u < 2; ++u) {
#pragma unroll
    for (int g = 0; g < 8; ++g) {
      int M   = u * 16 + g + h * 8;  // row per C/D VGPR layout
      int n   = wBase + M;
      int idx = bidx[u][g];
      float4 cb = *reinterpret_cast<const float4*>(emb + (size_t)idx * EDIM + k0);
      float4 xv = *reinterpret_cast<const float4*>(x   + (size_t)n   * EDIM + k0);
      *reinterpret_cast<float4*>(out + XQ_OFF + (size_t)n * EDIM + k0) = cb;
      *reinterpret_cast<float4*>(out + CB_OFF + (size_t)n * EDIM + k0) = cb;
      float dx = cb.x - xv.x, dy = cb.y - xv.y, dz = cb.z - xv.z, dw = cb.w - xv.w;
      lossAcc += dx*dx + dy*dy + dz*dz + dw*dw;
      if (rloc == 0) out[IDX_OFF + n] = (float)idx;
    }
  }

  // deterministic per-wave loss partial (no fp atomics)
#pragma unroll
  for (int mask = 1; mask < 32; mask <<= 1) lossAcc += __shfl_xor(lossAcc, mask, 32);
  if (lane == 0) partial[blockIdx.x * 4 + wave] = lossAcc;
}

// ---------------------------------------------------------------------------
// Kernel 3: deterministic final loss reduction (1024 partials).
// ---------------------------------------------------------------------------
__global__ __launch_bounds__(256) void vq_loss(
    const float* __restrict__ partial, float* __restrict__ out) {
  __shared__ float red[256];
  float s = 0.f;
  for (int i = threadIdx.x; i < 1024; i += 256) s += partial[i];
  red[threadIdx.x] = s;
  __syncthreads();
  if (threadIdx.x == 0) {
    float t = 0.f;
    for (int i = 0; i < 256; ++i) t += red[i];
    out[LOSS_OFF] = t * LOSS_SCALE;
  }
}

extern "C" void kernel_launch(void* const* d_in, const int* in_sizes, int n_in,
                              void* d_out, int out_size, void* d_ws, size_t ws_size,
                              hipStream_t stream) {
  const float* x   = (const float*)d_in[0];   // (1024,32,64) f32
  const float* emb = (const float*)d_in[1];   // (8192,64)    f32
  float* out = (float*)d_out;

  _Float16* wsB = (_Float16*)d_ws;                              // 1 MB B fragments
  float* partial = (float*)((char*)d_ws + (size_t)N_E * EDIM * sizeof(_Float16));

  vq_prep_codebook<<<NTILE / 4, 128, 0, stream>>>(emb, wsB);    // 512 waves
  vq_main<<<N_LAT / 128, 128, 0, stream>>>(x, emb, wsB, out, partial);
  vq_loss<<<1, 256, 0, stream>>>(partial, out);
}
